// MultiHeadDotProductAttention_70205535421358
// MI455X (gfx1250) — matrix operand order repaired and checked by reading, not compiled, not run on hardware
//
#include <hip/hip_runtime.h>
#include <cstdint>

// ---------------- CDNA5 WMMA types & helpers -------------------------------
typedef __attribute__((ext_vector_type(16))) __bf16 v16bf;
typedef __attribute__((ext_vector_type(8)))  float  v8f;

union AB16 { v16bf v; uint4 q[2]; };

__device__ inline v8f zero8() {
  v8f z = {0.f,0.f,0.f,0.f,0.f,0.f,0.f,0.f};
  return z;
}

__device__ inline v8f wmma_bf16(v16bf a, v16bf b, v8f c) {
  // D(f32 16x16) = A(bf16 16x32) * B(bf16 32x16) + C
  return __builtin_amdgcn_wmma_f32_16x16x32_bf16(
      /*neg_a=*/false, a, /*neg_b=*/false, b,
      /*c_mod=*/(short)0, c, /*reuse_a=*/false, /*reuse_b=*/false);
}

// A fragment 16x32 bf16 from row-major LDS tile [16][stride] (K contiguous).
// lane: row = lane&15; k-base = (lane>>4)*8; holds k..k+7 and k+16..k+23.
__device__ inline v16bf frag_a(const __bf16* base, int stride, int lane) {
  int r = lane & 15, kb = (lane >> 4) * 8;
  AB16 u;
  u.q[0] = *(const uint4*)(base + r * stride + kb);
  u.q[1] = *(const uint4*)(base + r * stride + kb + 16);
  return u.v;
}

// B fragment 32x16 bf16 from LDS tile stored as B^T: [N][stride] (K contiguous).
// lane: col = lane&15; k-base = (lane>>4)*8; holds k..k+7 and k+16..k+23.
__device__ inline v16bf frag_b(const __bf16* base, int stride, int lane) {
  int n = lane & 15, kb = (lane >> 4) * 8;
  AB16 u;
  u.q[0] = *(const uint4*)(base + n * stride + kb);
  u.q[1] = *(const uint4*)(base + n * stride + kb + 16);
  return u.v;
}

// ---------------- CDNA5 async global->LDS path -----------------------------
__device__ inline uint32_t lds_off_of(const void* p) {
  return (uint32_t)(uintptr_t)p;   // low 32 bits of LDS-aperture addr = offset
}

__device__ inline void async_cp16(uint32_t lds_off, const void* sbase, uint32_t goff) {
  // per-lane 16B copy global -> LDS, tracked by ASYNCcnt (GVS addressing)
  asm volatile("global_load_async_to_lds_b128 %0, %1, %2"
               :: "v"(lds_off), "v"(goff),
                  "s"((unsigned long long)(uintptr_t)sbase)
               : "memory");
}

__device__ inline void wait_async0() {
  asm volatile("s_wait_asynccnt 0" ::: "memory");
}
// wait until <=4 outstanding: previous tile's 4 copies done, next tile in flight
__device__ inline void wait_async4() {
  asm volatile("s_wait_asynccnt 4" ::: "memory");
}

// round-to-nearest-even fp32 -> bf16
__device__ inline __bf16 f2bf(float f) {
  union { float f; uint32_t u; } a; a.f = f;
  uint32_t r = a.u + 0x7FFFu + ((a.u >> 16) & 1u);
  union { unsigned short s; __bf16 b; } o; o.s = (unsigned short)(r >> 16);
  return o.b;
}

// ---------------- Kernel 1a: fp32 -> bf16 conversion -----------------------
__global__ __launch_bounds__(256)
void conv_f32_bf16(const float* __restrict__ in, __bf16* __restrict__ out, int n) {
  int i = (blockIdx.x * 256 + threadIdx.x) * 4;
  if (i >= n) return;
  float4 f = *(const float4*)(in + i);
  out[i + 0] = f2bf(f.x);
  out[i + 1] = f2bf(f.y);
  out[i + 2] = f2bf(f.z);
  out[i + 3] = f2bf(f.w);
}

// ---------------- Kernel 1b: fp32 [K][N] -> bf16 [N][K] transpose ----------
// Writes coalesced; strided reads hit L2 (whole matrix = 16MB << 192MB L2).
__global__ __launch_bounds__(256)
void conv_f32_bf16_t(const float* __restrict__ in, __bf16* __restrict__ out,
                     int K, int N) {
  int o = blockIdx.x * 256 + threadIdx.x;
  if (o >= K * N) return;
  int n = o / K, k = o - n * K;
  out[o] = f2bf(in[(size_t)k * N + n]);
}

// ---------------- Kernel 2: bf16 GEMM with epilogues -----------------------
// C[M,N] = A[M,K] * W[K,N], W given PRE-TRANSPOSED as Wt[N][K].
// Block tile 128x128, K-step 32, 256 thr = 8 waves (4 x 2).
// Double-buffered fully-async LDS staging (4 async copies / thread / tile).
// mode 0: RMS-norm over 128-wide N tile (one head) * scale -> bf16 [M][N]
// mode 1: bf16 store transposed as V^T [B][H][D][S]   (S = 2048 fixed)
// mode 2: f32 store [M][N] (final output projection)
#define GEMM_M 128
#define GEMM_N 128
#define GEMM_K 32
#define XST 40   // LDS row stride (elements): 32 + 8 pad, 80B (16B aligned)
#define WST 40
#define SEQ 2048

__global__ __launch_bounds__(256)
void gemm_bf16(const __bf16* __restrict__ A, const __bf16* __restrict__ Wt,
               void* __restrict__ outp, const float* __restrict__ scale,
               int M, int K, int N, int mode)
{
  __shared__ __bf16 sX[2][GEMM_M * XST];   // 2 x 10240 B
  __shared__ __bf16 sW[2][GEMM_N * WST];   // 2 x 10240 B
  __shared__ float  sSq[GEMM_M * 2];       // per-row ssq partials (per wn half)
  __shared__ float  sInv[GEMM_M];

  const int tid  = threadIdx.x;
  const int lane = tid & 31;
  const int wave = tid >> 5;
  const int wm   = wave & 3;    // 4 waves along M (32 rows each)
  const int wn   = wave >> 2;   // 2 waves along N (64 cols each)
  const int m0   = blockIdx.y * GEMM_M;
  const int n0   = blockIdx.x * GEMM_N;

  v8f acc[2][4];
  #pragma unroll
  for (int mi = 0; mi < 2; ++mi)
    #pragma unroll
    for (int j = 0; j < 4; ++j) acc[mi][j] = zero8();

  // staging map: 128 rows x 32 k, 16 elems (2x16B) per thread per matrix
  const int srow = tid >> 1, spart = (tid & 1) * 16;

  auto stage = [&](int k0, int buf) {
    uint32_t lx = lds_off_of(&sX[buf][srow * XST + spart]);
    uint32_t gx = (uint32_t)(((size_t)(m0 + srow) * K + k0 + spart) * 2);
    async_cp16(lx,      A, gx);
    async_cp16(lx + 16, A, gx + 16);
    uint32_t lw = lds_off_of(&sW[buf][srow * WST + spart]);
    uint32_t gw = (uint32_t)(((size_t)(n0 + srow) * K + k0 + spart) * 2);
    async_cp16(lw,      Wt, gw);
    async_cp16(lw + 16, Wt, gw + 16);
  };

  stage(0, 0);
  const int nk = K / GEMM_K;
  for (int i = 0; i < nk; ++i) {
    const int cur = i & 1;
    if (i + 1 < nk) {
      stage((i + 1) * GEMM_K, cur ^ 1);  // prefetch next tile (stays in flight)
      wait_async4();                     // previous tile's 4 copies complete
    } else {
      wait_async0();
    }
    __syncthreads();

    v16bf af0 = frag_a(&sX[cur][(wm * 32) * XST],        XST, lane);
    v16bf af1 = frag_a(&sX[cur][(wm * 32 + 16) * XST],   XST, lane);
    #pragma unroll
    for (int j = 0; j < 4; ++j) {
      v16bf bf = frag_b(&sW[cur][(wn * 64 + j * 16) * WST], WST, lane);
      acc[0][j] = wmma_bf16(af0, bf, acc[0][j]);
      acc[1][j] = wmma_bf16(af1, bf, acc[1][j]);
    }
    __syncthreads();
  }

  // C/D layout: per tile, row = rhalf+g, col = cl
  const int rhalf = (lane >> 4) * 8;
  const int cl = lane & 15;

  if (mode == 2) {
    float* o = (float*)outp;
    #pragma unroll
    for (int mi = 0; mi < 2; ++mi)
      #pragma unroll
      for (int j = 0; j < 4; ++j) {
        int c = n0 + wn * 64 + j * 16 + cl;
        #pragma unroll
        for (int g = 0; g < 8; ++g)
          o[(size_t)(m0 + wm * 32 + mi * 16 + rhalf + g) * N + c] = acc[mi][j][g];
      }
  } else if (mode == 1) {
    // store V transposed: vt[b][h][d][s], b = m/SEQ, s = m%SEQ, h = c/128, d = c%128
    __bf16* o = (__bf16*)outp;
    const int heads = N >> 7;
    #pragma unroll
    for (int mi = 0; mi < 2; ++mi)
      #pragma unroll
      for (int j = 0; j < 4; ++j) {
        int c = n0 + wn * 64 + j * 16 + cl;
        int h = c >> 7, d = c & 127;
        #pragma unroll
        for (int g = 0; g < 8; ++g) {
          int m = m0 + wm * 32 + mi * 16 + rhalf + g;
          size_t idx = (((size_t)(m / SEQ) * heads + h) * 128 + d) * SEQ + (m % SEQ);
          o[idx] = f2bf(acc[mi][j][g]);
        }
      }
  } else {
    // RMS over the 128-wide head slice: shuffle-reduce 64-col partials per wave,
    // combine the two wn halves through 1KB of LDS.
    #pragma unroll
    for (int mi = 0; mi < 2; ++mi) {
      float ssq[8];
      #pragma unroll
      for (int g = 0; g < 8; ++g) {
        float s = 0.f;
        #pragma unroll
        for (int j = 0; j < 4; ++j) { float v = acc[mi][j][g]; s += v * v; }
        ssq[g] = s;
      }
      #pragma unroll
      for (int msk = 1; msk < 16; msk <<= 1)
        #pragma unroll
        for (int g = 0; g < 8; ++g)
          ssq[g] += __shfl_xor(ssq[g], msk, 32);
      #pragma unroll
      for (int g = 0; g < 8; ++g)
        if (cl == g)
          sSq[(wm * 32 + mi * 16 + rhalf + g) * 2 + wn] = ssq[g];
    }
    __syncthreads();
    if (tid < GEMM_M)
      sInv[tid] = rsqrtf((sSq[tid * 2] + sSq[tid * 2 + 1]) * (1.0f / 128.0f) + 1e-6f);
    __syncthreads();
    __bf16* o = (__bf16*)outp;
    #pragma unroll
    for (int mi = 0; mi < 2; ++mi)
      #pragma unroll
      for (int j = 0; j < 4; ++j) {
        int c = wn * 64 + j * 16 + cl;      // col within head (== scale index)
        float sc = scale[c];
        #pragma unroll
        for (int g = 0; g < 8; ++g) {
          int r = wm * 32 + mi * 16 + rhalf + g;
          o[(size_t)(m0 + r) * N + n0 + c] = f2bf(acc[mi][j][g] * sInv[r] * sc);
        }
      }
  }
}

// ---------------- Kernel 3: flash attention per (b, h, 128-q-tile) ---------
// Q/K read from [B*S][H*D] bf16; V read PRE-TRANSPOSED as vt[B][H][D][S].
#define AQ 128    // q rows per block (8 waves x 16 rows)
#define AK 32     // keys per iteration
#define QST 136   // sQ row stride (128 + 8 pad), 272B
#define KST2 136
#define VST 40    // sV (V^T [d][key]) row stride: 32 + 8 pad
#define PST 40    // per-wave probability tile stride

__global__ __launch_bounds__(256)
void attn_fwd(const __bf16* __restrict__ Q, const __bf16* __restrict__ Kmat,
              const __bf16* __restrict__ Vt, __bf16* __restrict__ O,
              int S, int HD)
{
  __shared__ __bf16 sQ[AQ * QST];        // 34816 B
  __shared__ __bf16 sK[AK * KST2];       //  8704 B
  __shared__ __bf16 sV[128 * VST];       // 10240 B (V^T: [d][key])
  __shared__ __bf16 sP[8 * 16 * PST];    // 10240 B (per-wave P tiles)

  const int tid = threadIdx.x, lane = tid & 31, wave = tid >> 5;
  const int qt = blockIdx.x, h = blockIdx.y, b = blockIdx.z;
  const int heads = HD >> 7;
  const size_t rowbase = (size_t)b * S + (size_t)qt * AQ;
  const __bf16* qptr  = Q    + rowbase * HD + h * 128;
  const __bf16* kptr  = Kmat + (size_t)b * S * HD + h * 128;
  const __bf16* vtptr = Vt   + ((size_t)b * heads + h) * 128 * S;

  // stage Q tile (128 x 128) once: 4 async 16B copies per thread
  {
    int r = tid >> 1, half = (tid & 1) * 64;
    #pragma unroll
    for (int c = 0; c < 4; ++c)
      async_cp16(lds_off_of(&sQ[r * QST + half + c * 16]), qptr,
                 (uint32_t)(((size_t)r * HD + half + c * 16) * 2));
  }

  v8f o_acc[8];
  #pragma unroll
  for (int j = 0; j < 8; ++j) o_acc[j] = zero8();
  float mrow[8], lrow[8];
  #pragma unroll
  for (int g = 0; g < 8; ++g) { mrow[g] = -1e30f; lrow[g] = 0.f; }
  const int rhalf = (lane >> 4) * 8;
  const int cl = lane & 15;

  for (int kb = 0; kb < S; kb += AK) {
    __syncthreads();
    // K tile 32x128: async (row-major == B^T layout for QK^T)
    {
      int r = tid >> 3, c = (tid & 7) * 16;
      uint32_t lo = lds_off_of(&sK[r * KST2 + c]);
      uint32_t go = (uint32_t)(((size_t)(kb + r) * HD + c) * 2);
      async_cp16(lo,      kptr, go);
      async_cp16(lo + 16, kptr, go + 16);
    }
    // V^T tile 128(d) x 32(key): async from vt (already [d][s])
    {
      int d = tid >> 1, hk = (tid & 1) * 16;
      uint32_t lo = lds_off_of(&sV[d * VST + hk]);
      uint32_t go = (uint32_t)(((size_t)d * S + kb + hk) * 2);
      async_cp16(lo,      vtptr, go);
      async_cp16(lo + 16, vtptr, go + 16);
    }
    wait_async0();
    __syncthreads();

    // scores S = Qn Kn^T : 16 rows x 32 keys per wave, d=128 in 4 WMMA steps
    v8f s0 = zero8(), s1 = zero8();
    #pragma unroll
    for (int kk = 0; kk < 4; ++kk) {
      v16bf aq = frag_a(&sQ[wave * 16 * QST + kk * 32], QST, lane);
      v16bf b0 = frag_b(&sK[0 * KST2 + kk * 32], KST2, lane);
      v16bf b1 = frag_b(&sK[16 * KST2 + kk * 32], KST2, lane);
      s0 = wmma_bf16(aq, b0, s0);
      s1 = wmma_bf16(aq, b1, s1);
    }

    // ---- online softmax (rows private to each wave; 16-lane reductions) ----
    float tmax[8];
    #pragma unroll
    for (int g = 0; g < 8; ++g) tmax[g] = fmaxf(s0[g], s1[g]);
    #pragma unroll
    for (int msk = 1; msk < 16; msk <<= 1) {
      #pragma unroll
      for (int g = 0; g < 8; ++g)
        tmax[g] = fmaxf(tmax[g], __shfl_xor(tmax[g], msk, 32));
    }
    float alpha[8], psum[8];
    #pragma unroll
    for (int g = 0; g < 8; ++g) {
      float mn = fmaxf(mrow[g], tmax[g]);
      alpha[g] = __expf(mrow[g] - mn);
      mrow[g] = mn;
    }
    #pragma unroll
    for (int g = 0; g < 8; ++g) {
      float p0 = __expf(s0[g] - mrow[g]);
      float p1 = __expf(s1[g] - mrow[g]);
      psum[g] = p0 + p1;
      // C layout -> A layout via per-wave LDS region (same-wave DS is in-order)
      sP[(wave * 16 + rhalf + g) * PST + cl]      = f2bf(p0);
      sP[(wave * 16 + rhalf + g) * PST + 16 + cl] = f2bf(p1);
    }
    #pragma unroll
    for (int msk = 1; msk < 16; msk <<= 1) {
      #pragma unroll
      for (int g = 0; g < 8; ++g)
        psum[g] += __shfl_xor(psum[g], msk, 32);
    }
    #pragma unroll
    for (int g = 0; g < 8; ++g) lrow[g] = lrow[g] * alpha[g] + psum[g];
    #pragma unroll
    for (int j = 0; j < 8; ++j) {
      #pragma unroll
      for (int g = 0; g < 8; ++g) o_acc[j][g] *= alpha[g];
    }

    // O += P @ V  (A: 16x32 probs, B: V^T[d][key])
    v16bf ap = frag_a(&sP[wave * 16 * PST], PST, lane);
    #pragma unroll
    for (int j = 0; j < 8; ++j) {
      v16bf bv = frag_b(&sV[(j * 16) * VST], VST, lane);
      o_acc[j] = wmma_bf16(ap, bv, o_acc[j]);
    }
  }

  // finalize: divide by running sum, store O[b,s,h,d] as bf16
  #pragma unroll
  for (int g = 0; g < 8; ++g) lrow[g] = 1.0f / lrow[g];
  #pragma unroll
  for (int j = 0; j < 8; ++j) {
    #pragma unroll
    for (int g = 0; g < 8; ++g) {
      size_t row = rowbase + (size_t)(wave * 16 + rhalf + g);
      O[row * HD + h * 128 + j * 16 + cl] = f2bf(o_acc[j][g] * lrow[g]);
    }
  }
}

// ---------------- host launch ----------------------------------------------
extern "C" void kernel_launch(void* const* d_in, const int* in_sizes, int n_in,
                              void* d_out, int out_size, void* d_ws, size_t ws_size,
                              hipStream_t stream) {
  (void)in_sizes; (void)n_in; (void)out_size; (void)ws_size;
  const int B = 2, S = 2048, E = 2048, H = 16, D = 128;
  const int M = B * S;      // 4096 rows
  const int HD = H * D;     // 2048

  const float* x       = (const float*)d_in[0];
  const float* wq      = (const float*)d_in[1];
  const float* wk      = (const float*)d_in[2];
  const float* wv      = (const float*)d_in[3];
  const float* wo      = (const float*)d_in[4];
  const float* q_scale = (const float*)d_in[5];
  const float* k_scale = (const float*)d_in[6];

  char* ws = (char*)d_ws;
  size_t off = 0;
  auto alloc = [&](size_t bytes) -> char* {
    char* p = ws + off;
    off += (bytes + 255) & ~(size_t)255;
    return p;
  };
  __bf16* xb   = (__bf16*)alloc((size_t)M * E * 2);    // x in bf16 [M][E]
  __bf16* wqt  = (__bf16*)alloc((size_t)E * HD * 2);   // wq^T [HD][E]
  __bf16* wkt  = (__bf16*)alloc((size_t)E * HD * 2);   // wk^T [HD][E]
  __bf16* wvt  = (__bf16*)alloc((size_t)E * HD * 2);   // wv^T [HD][E]
  __bf16* wot  = (__bf16*)alloc((size_t)HD * E * 2);   // wo^T [E][HD]
  __bf16* qn   = (__bf16*)alloc((size_t)M * HD * 2);   // RMS-normed Q [M][HD]
  __bf16* kn   = (__bf16*)alloc((size_t)M * HD * 2);   // RMS-normed K [M][HD]
  __bf16* vt   = (__bf16*)alloc((size_t)M * HD * 2);   // V^T [B][H][D][S]
  __bf16* ob   = (__bf16*)alloc((size_t)M * HD * 2);   // attention out [M][HD]

  conv_f32_bf16<<<dim3((M * E / 4 + 255) / 256), 256, 0, stream>>>(x, xb, M * E);
  dim3 tgrid((E * HD + 255) / 256);
  conv_f32_bf16_t<<<tgrid, 256, 0, stream>>>(wq, wqt, E, HD);
  conv_f32_bf16_t<<<tgrid, 256, 0, stream>>>(wk, wkt, E, HD);
  conv_f32_bf16_t<<<tgrid, 256, 0, stream>>>(wv, wvt, E, HD);
  conv_f32_bf16_t<<<tgrid, 256, 0, stream>>>(wo, wot, HD, E);

  dim3 ggrid(HD / GEMM_N, M / GEMM_M);  // (16, 32)
  gemm_bf16<<<ggrid, 256, 0, stream>>>(xb, wqt, qn, q_scale, M, E, HD, 0);
  gemm_bf16<<<ggrid, 256, 0, stream>>>(xb, wkt, kn, k_scale, M, E, HD, 0);
  gemm_bf16<<<ggrid, 256, 0, stream>>>(xb, wvt, vt, nullptr, M, E, HD, 1);

  attn_fwd<<<dim3(S / AQ, H, B), 256, 0, stream>>>(qn, kn, vt, ob, S, HD);

  gemm_bf16<<<dim3(E / GEMM_N, M / GEMM_M), 256, 0, stream>>>(
      ob, wot, d_out, nullptr, M, HD, E, 2);
}
